// NextBasketEncoder_81226421502137
// MI455X (gfx1250) — matrix operand order repaired and checked by reading, not compiled, not run on hardware
//
#include <hip/hip_runtime.h>

// ---------------------------------------------------------------------------
// CDNA5 WMMA helpers (wave32, v_wmma_f32_16x16x32_f16)
// ---------------------------------------------------------------------------
typedef __attribute__((ext_vector_type(16))) _Float16 v16h;
typedef __attribute__((ext_vector_type(8)))  float    v8f;

__device__ __forceinline__ v8f wmma16(v16h a, v16h b, v8f c) {
  return __builtin_amdgcn_wmma_f32_16x16x32_f16(
      /*neg_a=*/false, a, /*neg_b=*/false, b,
      /*c_mod=*/(short)0, c, /*reuse_a=*/false, /*reuse_b=*/false);
}

// A fragment (16x32 f16) from row-major f16 source with leading dim ld.
// Lane l: m = l&15, khalf = l>>4.  k = 16*khalf+e.
__device__ __forceinline__ v16h ld_a(const _Float16* S, int ld) {
  int l = threadIdx.x & 31;
  int m = l & 15, kh = (l >> 4) * 16;
  v16h a;
#pragma unroll
  for (int e = 0; e < 16; ++e) {
    int k = kh + e;
    a[e] = S[m * ld + k];
  }
  return a;
}

// B fragment from packed fragment-major buffer (1KB per fragment).
__device__ __forceinline__ v16h ld_b_packed(const _Float16* P, int frag) {
  return *(const v16h*)(P + (size_t)frag * 512 + (size_t)(threadIdx.x & 31) * 16);
}

// B fragment from row-major KxN f16 source: B[k][n] = S[k*ld + n]
__device__ __forceinline__ v16h ld_b_kn(const _Float16* S, int ld) {
  int l = threadIdx.x & 31;
  int n = l & 15, kb = (l >> 4) * 16;
  v16h b;
#pragma unroll
  for (int e = 0; e < 16; ++e) b[e] = S[(kb + e) * ld + n];
  return b;
}

// B fragment from row-major NxK f16 source (implicit transpose): B[k][n] = S[n*ld + k]
__device__ __forceinline__ v16h ld_b_nk(const _Float16* S, int ld) {
  int l = threadIdx.x & 31;
  int n = l & 15, kb = (l >> 4) * 16;
  v16h b;
#pragma unroll
  for (int e = 0; e < 16; ++e) b[e] = S[n * ld + kb + e];
  return b;
}

// ---------------------------------------------------------------------------
// Weight pack: W [N][K] row-major f32  ->  f16 fragments, frag = kt*(N/16)+nt,
// element: out[frag*512 + lane*16 + e] = W[nt*16 + (lane&15)][kt*32 + (lane>>4)*16 + e]
// ---------------------------------------------------------------------------
__global__ void pack_w(const float* __restrict__ W, _Float16* __restrict__ out,
                       int N, int K) {
  int ntiles = N / 16;
  int nfrag = (K / 32) * ntiles;
  int tid = blockIdx.x * blockDim.x + threadIdx.x;
  if (tid >= nfrag * 32) return;
  int f = tid >> 5, lane = tid & 31;
  int kt = f / ntiles, nt = f % ntiles;
  int n = nt * 16 + (lane & 15);
  int kb = kt * 32 + ((lane >> 4) * 16);
  _Float16* o = out + (size_t)f * 512 + (size_t)lane * 16;
#pragma unroll
  for (int e = 0; e < 16; ++e) o[e] = (_Float16)W[(size_t)n * K + kb + e];
}

// ---------------------------------------------------------------------------
// K1: per-order item embedding gather + 4-head MHA (L=20, hd=16) + mean pool
//     + output projection  ->  order_emb [B*O, 64]
// block = 1 order, 128 threads (4 waves)
// ---------------------------------------------------------------------------
__global__ void __launch_bounds__(128) k_item_attn(
    const int* __restrict__ hist, const float* __restrict__ emb,
    const _Float16* __restrict__ Wqkv, const float* __restrict__ bqkv,
    const float* __restrict__ Wout, const float* __restrict__ bout,
    float* __restrict__ order_emb) {
  __shared__ _Float16 Xh[32 * 64];
  __shared__ float QKV[32 * 192];
  __shared__ float AO[20 * 64];
  __shared__ float PL[64];
  int ord = blockIdx.x;
  const int* idx = hist + (size_t)ord * 20;
  int tid = threadIdx.x, wv = tid >> 5;

  for (int i = tid; i < 32 * 64; i += 128) {
    int r = i >> 6, c = i & 63;
    float v = 0.f;
    if (r < 20) { int it = idx[r]; v = emb[(size_t)it * 64 + c]; }
    Xh[i] = (_Float16)v;
  }
  __syncthreads();

  // QKV projection: M=32(pad), N=192, K=64  -> 2x12 tiles, 2 k-frags
  for (int t = wv; t < 24; t += 4) {
    int mt = t / 12, nt = t % 12;
    v8f c = {};
#pragma unroll
    for (int kf = 0; kf < 2; ++kf) {
      v16h a = ld_a(Xh + mt * 16 * 64 + kf * 32, 64);
      v16h b = ld_b_packed(Wqkv, kf * 12 + nt);
      c = wmma16(a, b, c);
    }
    int l = tid & 31, n = l & 15, mb = (l >> 4) * 8;
    float bb = bqkv[nt * 16 + n];
#pragma unroll
    for (int r = 0; r < 8; ++r)
      QKV[(mt * 16 + mb + r) * 192 + nt * 16 + n] = c[r] + bb;
  }
  __syncthreads();

  // attention: 80 threads, one (head, query) each
  if (tid < 80) {
    int h = tid / 20, q = tid % 20;
    float s[20], mx = -1e30f;
    for (int k = 0; k < 20; ++k) {
      float d = 0.f;
#pragma unroll
      for (int j = 0; j < 16; ++j)
        d += QKV[q * 192 + h * 16 + j] * QKV[k * 192 + 64 + h * 16 + j];
      s[k] = d * 0.25f;
      mx = fmaxf(mx, s[k]);
    }
    float sum = 0.f;
    for (int k = 0; k < 20; ++k) { s[k] = __expf(s[k] - mx); sum += s[k]; }
    float inv = 1.f / sum;
#pragma unroll
    for (int j = 0; j < 16; ++j) {
      float o = 0.f;
      for (int k = 0; k < 20; ++k) o += s[k] * QKV[k * 192 + 128 + h * 16 + j];
      AO[q * 64 + h * 16 + j] = o * inv;
    }
  }
  __syncthreads();
  if (tid < 64) {
    float acc = 0.f;
    for (int r = 0; r < 20; ++r) acc += AO[r * 64 + tid];
    PL[tid] = acc * (1.f / 20.f);
  }
  __syncthreads();
  if (tid < 64) {   // mean(A W^T)+b == mean(A) W^T + b
    float acc = bout[tid];
    for (int i = 0; i < 64; ++i) acc += PL[i] * Wout[tid * 64 + i];
    order_emb[(size_t)ord * 64 + tid] = acc;
  }
}

// ---------------------------------------------------------------------------
// K2: bidirectional GRU over O=50 steps; one block = 16 batch rows, one dir.
// gi = x@wi^T (24 ntiles, K=64), gh = h@wh^T (24 ntiles, K=128), gates VALU.
// ---------------------------------------------------------------------------
__global__ void __launch_bounds__(256) k_gru(
    const float* __restrict__ oe,
    const _Float16* __restrict__ Pwi_f, const _Float16* __restrict__ Pwh_f,
    const float* __restrict__ bi_f, const float* __restrict__ bh_f,
    const _Float16* __restrict__ Pwi_b, const _Float16* __restrict__ Pwh_b,
    const float* __restrict__ bi_b, const float* __restrict__ bh_b,
    float* __restrict__ g) {
  __shared__ _Float16 Xh[16 * 64];
  __shared__ _Float16 Hh[16 * 128];
  __shared__ float GI[16 * 384];
  __shared__ float GH[16 * 384];
  __shared__ float Hc[16 * 128];
  int d = blockIdx.y;
  int b0 = blockIdx.x * 16;
  const _Float16* Pwi = d ? Pwi_b : Pwi_f;
  const _Float16* Pwh = d ? Pwh_b : Pwh_f;
  const float* bi = d ? bi_b : bi_f;
  const float* bh = d ? bh_b : bh_f;
  int tid = threadIdx.x, wv = tid >> 5;

  for (int i = tid; i < 16 * 128; i += 256) Hc[i] = 0.f;
  __syncthreads();

  for (int t = 0; t < 50; ++t) {
    int tt = d ? (49 - t) : t;
    for (int i = tid; i < 16 * 64; i += 256) {
      int r = i >> 6, c = i & 63;
      Xh[i] = (_Float16)oe[((size_t)(b0 + r) * 50 + tt) * 64 + c];
    }
    for (int i = tid; i < 16 * 128; i += 256) Hh[i] = (_Float16)Hc[i];
    __syncthreads();

    for (int nt = wv; nt < 24; nt += 8) {
      v8f ci = {};
#pragma unroll
      for (int kf = 0; kf < 2; ++kf) {
        v16h a = ld_a(Xh + kf * 32, 64);
        v16h b = ld_b_packed(Pwi, kf * 24 + nt);
        ci = wmma16(a, b, ci);
      }
      v8f ch = {};
#pragma unroll
      for (int kf = 0; kf < 4; ++kf) {
        v16h a = ld_a(Hh + kf * 32, 128);
        v16h b = ld_b_packed(Pwh, kf * 24 + nt);
        ch = wmma16(a, b, ch);
      }
      int l = tid & 31, n = l & 15, mb = (l >> 4) * 8;
      float vbi = bi[nt * 16 + n], vbh = bh[nt * 16 + n];
#pragma unroll
      for (int r = 0; r < 8; ++r) {
        GI[(mb + r) * 384 + nt * 16 + n] = ci[r] + vbi;
        GH[(mb + r) * 384 + nt * 16 + n] = ch[r] + vbh;
      }
    }
    __syncthreads();

    for (int i = tid; i < 16 * 128; i += 256) {
      int r = i >> 7, c = i & 127;
      float ir = GI[r * 384 + c], iz = GI[r * 384 + 128 + c], in_ = GI[r * 384 + 256 + c];
      float hr = GH[r * 384 + c], hz = GH[r * 384 + 128 + c], hn = GH[r * 384 + 256 + c];
      float rr = 1.f / (1.f + __expf(-(ir + hr)));
      float zz = 1.f / (1.f + __expf(-(iz + hz)));
      float nn = tanhf(in_ + rr * hn);
      float h2 = (1.f - zz) * nn + zz * Hc[i];
      Hc[i] = h2;
      g[((size_t)(b0 + r) * 50 + tt) * 256 + d * 128 + c] = h2;
    }
    __syncthreads();
  }
}

// ---------------------------------------------------------------------------
// K3a: transformer QKV GEMM  [51200,256] @ [256,768] -> f16 qkv
// ---------------------------------------------------------------------------
__global__ void __launch_bounds__(256) k_t_qkv(
    const float* __restrict__ g, const _Float16* __restrict__ Pw,
    const float* __restrict__ bias, _Float16* __restrict__ qkv) {
  __shared__ _Float16 Xh[16 * 256];
  int row0 = blockIdx.x * 16;
  int tid = threadIdx.x, wv = tid >> 5;
  for (int i = tid; i < 16 * 256; i += 256) Xh[i] = (_Float16)g[(size_t)row0 * 256 + i];
  __syncthreads();
  for (int nt = wv; nt < 48; nt += 8) {
    v8f c = {};
#pragma unroll
    for (int kf = 0; kf < 8; ++kf) {
      v16h a = ld_a(Xh + kf * 32, 256);
      v16h b = ld_b_packed(Pw, kf * 48 + nt);
      c = wmma16(a, b, c);
    }
    int l = tid & 31, n = l & 15, mb = (l >> 4) * 8;
    float vb = bias[nt * 16 + n];
#pragma unroll
    for (int r = 0; r < 8; ++r)
      qkv[((size_t)row0 + mb + r) * 768 + nt * 16 + n] = (_Float16)(c[r] + vb);
  }
}

// ---------------------------------------------------------------------------
// K3b: per-(batch, head) attention over S=50 (padded to 64), hd=64
// ---------------------------------------------------------------------------
__global__ void __launch_bounds__(256) k_t_attn(
    const _Float16* __restrict__ qkv, _Float16* __restrict__ ao) {
  __shared__ _Float16 Qh[64 * 64];
  __shared__ _Float16 Kh[64 * 64];   // reused for softmax probs
  __shared__ _Float16 Vh[64 * 64];
  __shared__ float SC[64 * 64];
  int b = blockIdx.x, h = blockIdx.y;
  int tid = threadIdx.x, wv = tid >> 5;
  const size_t base = (size_t)b * 50 * 768;

  for (int i = tid; i < 64 * 64; i += 256) {
    int s = i >> 6, j = i & 63;
    _Float16 q = (_Float16)0.f, k = (_Float16)0.f, v = (_Float16)0.f;
    if (s < 50) {
      const _Float16* p = qkv + base + (size_t)s * 768 + h * 64 + j;
      q = p[0]; k = p[256]; v = p[512];
    }
    Qh[i] = q; Kh[i] = k; Vh[i] = v;
  }
  __syncthreads();

  for (int t = wv; t < 16; t += 8) {          // scores = Q K^T * 0.125
    int mt = t >> 2, nt = t & 3;
    v8f c = {};
#pragma unroll
    for (int kf = 0; kf < 2; ++kf) {
      v16h a = ld_a(Qh + mt * 16 * 64 + kf * 32, 64);
      v16h bb = ld_b_nk(Kh + nt * 16 * 64 + kf * 32, 64);
      c = wmma16(a, bb, c);
    }
    int l = tid & 31, n = l & 15, mb = (l >> 4) * 8;
#pragma unroll
    for (int r = 0; r < 8; ++r)
      SC[(mt * 16 + mb + r) * 64 + nt * 16 + n] = c[r] * 0.125f;
  }
  __syncthreads();

  if (tid < 64) {                              // row softmax (cols 0..49)
    int r = tid;
    float mx = -1e30f;
    for (int c = 0; c < 50; ++c) mx = fmaxf(mx, SC[r * 64 + c]);
    float sum = 0.f;
    for (int c = 0; c < 50; ++c) sum += __expf(SC[r * 64 + c] - mx);
    float inv = 1.f / sum;
    for (int c = 0; c < 64; ++c)
      Kh[r * 64 + c] = (_Float16)((c < 50) ? __expf(SC[r * 64 + c] - mx) * inv : 0.f);
  }
  __syncthreads();

  for (int t = wv; t < 16; t += 8) {          // out = P V
    int mt = t >> 2, nt = t & 3;
    v8f c = {};
#pragma unroll
    for (int kf = 0; kf < 2; ++kf) {
      v16h a = ld_a(Kh + mt * 16 * 64 + kf * 32, 64);
      v16h bb = ld_b_kn(Vh + (kf * 32) * 64 + nt * 16, 64);
      c = wmma16(a, bb, c);
    }
    int l = tid & 31, n = l & 15, mb = (l >> 4) * 8;
#pragma unroll
    for (int r = 0; r < 8; ++r) {
      int m = mt * 16 + mb + r;
      if (m < 50)
        ao[((size_t)b * 50 + m) * 256 + h * 64 + nt * 16 + n] = (_Float16)c[r];
    }
  }
}

// ---------------------------------------------------------------------------
// shared LayerNorm over 16 rows x 256 cols staged in LDS (256 threads)
// ---------------------------------------------------------------------------
__device__ __forceinline__ void ln_rows16x256(
    const float* __restrict__ R, float* __restrict__ part, float* __restrict__ red,
    const float* __restrict__ lng, const float* __restrict__ lnb,
    float* __restrict__ out, size_t row0) {
  int tid = threadIdx.x;
  int r = tid >> 4, ln = tid & 15;
  float s = 0.f;
  for (int c = ln; c < 256; c += 16) s += R[r * 256 + c];
  part[tid] = s;
  __syncthreads();
  if (ln == 0) {
    float tot = 0.f;
    for (int i = 0; i < 16; ++i) tot += part[r * 16 + i];
    red[r] = tot * (1.f / 256.f);
  }
  __syncthreads();
  float mean = red[r];
  float vs = 0.f;
  for (int c = ln; c < 256; c += 16) { float d = R[r * 256 + c] - mean; vs += d * d; }
  __syncthreads();
  part[tid] = vs;
  __syncthreads();
  if (ln == 0) {
    float tot = 0.f;
    for (int i = 0; i < 16; ++i) tot += part[r * 16 + i];
    red[r] = tot * (1.f / 256.f);
  }
  __syncthreads();
  float inv = rsqrtf(red[r] + 1e-5f);
  for (int c = ln; c < 256; c += 16)
    out[(row0 + r) * 256 + c] = (R[r * 256 + c] - mean) * inv * lng[c] + lnb[c];
}

// ---------------------------------------------------------------------------
// K3c: attn-out projection + residual + LN1  -> h1
// ---------------------------------------------------------------------------
__global__ void __launch_bounds__(256) k_t_proj_ln(
    const _Float16* __restrict__ ao, const _Float16* __restrict__ Pw,
    const float* __restrict__ bias, const float* __restrict__ res,
    const float* __restrict__ lng, const float* __restrict__ lnb,
    float* __restrict__ out) {
  __shared__ _Float16 Xh[16 * 256];
  __shared__ float R[16 * 256];
  __shared__ float part[256];
  __shared__ float red[16];
  int row0 = blockIdx.x * 16;
  int tid = threadIdx.x, wv = tid >> 5;
  for (int i = tid; i < 16 * 256; i += 256) Xh[i] = ao[(size_t)row0 * 256 + i];
  __syncthreads();
  for (int nt = wv; nt < 16; nt += 8) {
    v8f c = {};
#pragma unroll
    for (int kf = 0; kf < 8; ++kf) {
      v16h a = ld_a(Xh + kf * 32, 256);
      v16h b = ld_b_packed(Pw, kf * 16 + nt);
      c = wmma16(a, b, c);
    }
    int l = tid & 31, n = l & 15, mb = (l >> 4) * 8;
    float vb = bias[nt * 16 + n];
#pragma unroll
    for (int r = 0; r < 8; ++r)
      R[(mb + r) * 256 + nt * 16 + n] =
          c[r] + vb + res[((size_t)row0 + mb + r) * 256 + nt * 16 + n];
  }
  __syncthreads();
  ln_rows16x256(R, part, red, lng, lnb, out, (size_t)row0);
}

// ---------------------------------------------------------------------------
// K3d: FFN (256->512 relu ->256) + residual + LN2 -> h2
// ---------------------------------------------------------------------------
__global__ void __launch_bounds__(256) k_t_ffn(
    const float* __restrict__ h1, const _Float16* __restrict__ Pw1,
    const float* __restrict__ b1, const _Float16* __restrict__ Pw2,
    const float* __restrict__ b2,
    const float* __restrict__ lng, const float* __restrict__ lnb,
    float* __restrict__ out) {
  __shared__ _Float16 Xh[16 * 256];
  __shared__ _Float16 U[16 * 512];
  __shared__ float R[16 * 256];
  __shared__ float part[256];
  __shared__ float red[16];
  int row0 = blockIdx.x * 16;
  int tid = threadIdx.x, wv = tid >> 5;
  for (int i = tid; i < 16 * 256; i += 256) Xh[i] = (_Float16)h1[(size_t)row0 * 256 + i];
  __syncthreads();
  for (int nt = wv; nt < 32; nt += 8) {       // FF1 + relu
    v8f c = {};
#pragma unroll
    for (int kf = 0; kf < 8; ++kf) {
      v16h a = ld_a(Xh + kf * 32, 256);
      v16h b = ld_b_packed(Pw1, kf * 32 + nt);
      c = wmma16(a, b, c);
    }
    int l = tid & 31, n = l & 15, mb = (l >> 4) * 8;
    float vb = b1[nt * 16 + n];
#pragma unroll
    for (int r = 0; r < 8; ++r)
      U[(mb + r) * 512 + nt * 16 + n] = (_Float16)fmaxf(c[r] + vb, 0.f);
  }
  __syncthreads();
  for (int nt = wv; nt < 16; nt += 8) {       // FF2 + residual
    v8f c = {};
#pragma unroll
    for (int kf = 0; kf < 16; ++kf) {
      v16h a = ld_a(U + kf * 32, 512);
      v16h b = ld_b_packed(Pw2, kf * 16 + nt);
      c = wmma16(a, b, c);
    }
    int l = tid & 31, n = l & 15, mb = (l >> 4) * 8;
    float vb = b2[nt * 16 + n];
#pragma unroll
    for (int r = 0; r < 8; ++r)
      R[(mb + r) * 256 + nt * 16 + n] =
          c[r] + vb + h1[((size_t)row0 + mb + r) * 256 + nt * 16 + n];
  }
  __syncthreads();
  ln_rows16x256(R, part, red, lng, lnb, out, (size_t)row0);
}

// ---------------------------------------------------------------------------
// K4: mean over O, temporal MLP, concat + combiner + LN + relu -> out [B,128]
// ---------------------------------------------------------------------------
__device__ __forceinline__ void ln_relu_lds(float* v, int n, const float* g,
                                            const float* bb, float* st) {
  if (threadIdx.x == 0) {
    float m = 0.f;
    for (int i = 0; i < n; ++i) m += v[i];
    m /= n;
    float var = 0.f;
    for (int i = 0; i < n; ++i) { float d = v[i] - m; var += d * d; }
    var /= n;
    st[0] = m; st[1] = rsqrtf(var + 1e-5f);
  }
  __syncthreads();
  if ((int)threadIdx.x < n) {
    int i = threadIdx.x;
    v[i] = fmaxf((v[i] - st[0]) * st[1] * g[i] + bb[i], 0.f);
  }
  __syncthreads();
}

__global__ void __launch_bounds__(256) k_final(
    const float* __restrict__ h2, const float* __restrict__ days_between,
    const float* __restrict__ days_since,
    const float* __restrict__ te1_w, const float* __restrict__ te1_b,
    const float* __restrict__ ln1g, const float* __restrict__ ln1b,
    const float* __restrict__ te2_w, const float* __restrict__ te2_b,
    const float* __restrict__ ln2g, const float* __restrict__ ln2b,
    const float* __restrict__ cw, const float* __restrict__ cb,
    const float* __restrict__ clng, const float* __restrict__ clnb,
    float* __restrict__ out) {
  __shared__ float sOE[256];
  __shared__ float tb[64];
  __shared__ float t2[64];
  __shared__ float ub[128];
  __shared__ float st[2];
  int b = blockIdx.x, tid = threadIdx.x;
  {
    float s = 0.f;
    for (int o = 0; o < 50; ++o) s += h2[((size_t)b * 50 + o) * 256 + tid];
    sOE[tid] = s * (1.f / 50.f);
  }
  if (tid == 0) {
    float s = 0.f;
    for (int i = 0; i < 10; ++i) s += days_between[b * 10 + i];
    st[0] = s * 0.1f; st[1] = days_since[b];
  }
  __syncthreads();
  float f0 = st[0], f1 = st[1];
  __syncthreads();
  if (tid < 32) tb[tid] = te1_b[tid] + f0 * te1_w[tid * 2] + f1 * te1_w[tid * 2 + 1];
  __syncthreads();
  ln_relu_lds(tb, 32, ln1g, ln1b, st);
  if (tid < 64) {
    float a = te2_b[tid];
    for (int i = 0; i < 32; ++i) a += tb[i] * te2_w[tid * 32 + i];
    t2[tid] = a;
  }
  __syncthreads();
  ln_relu_lds(t2, 64, ln2g, ln2b, st);
  if (tid < 128) {
    float a = cb[tid];
    for (int i = 0; i < 256; ++i) a += sOE[i] * cw[tid * 320 + i];
    for (int i = 0; i < 64; ++i) a += t2[i] * cw[tid * 320 + 256 + i];
    ub[tid] = a;
  }
  __syncthreads();
  ln_relu_lds(ub, 128, clng, clnb, st);
  if (tid < 128) out[(size_t)b * 128 + tid] = ub[tid];
}

// ---------------------------------------------------------------------------
// host launcher
// ---------------------------------------------------------------------------
extern "C" void kernel_launch(void* const* d_in, const int* in_sizes, int n_in,
                              void* d_out, int out_size, void* d_ws, size_t ws_size,
                              hipStream_t stream) {
  (void)in_sizes; (void)n_in; (void)out_size; (void)ws_size;
  const int*   hist        = (const int*)d_in[0];
  const float* daysb       = (const float*)d_in[1];
  const float* dayss       = (const float*)d_in[2];
  const float* item_emb    = (const float*)d_in[3];
  const float* ia_in_w     = (const float*)d_in[4];
  const float* ia_in_b     = (const float*)d_in[5];
  const float* ia_out_w    = (const float*)d_in[6];
  const float* ia_out_b    = (const float*)d_in[7];
  const float* gf_wi       = (const float*)d_in[8];
  const float* gf_wh       = (const float*)d_in[9];
  const float* gf_bi       = (const float*)d_in[10];
  const float* gf_bh       = (const float*)d_in[11];
  const float* gb_wi       = (const float*)d_in[12];
  const float* gb_wh       = (const float*)d_in[13];
  const float* gb_bi       = (const float*)d_in[14];
  const float* gb_bh       = (const float*)d_in[15];
  const float* ta_in_w     = (const float*)d_in[16];
  const float* ta_in_b     = (const float*)d_in[17];
  const float* ta_out_w    = (const float*)d_in[18];
  const float* ta_out_b    = (const float*)d_in[19];
  const float* t_ln1_g     = (const float*)d_in[20];
  const float* t_ln1_b     = (const float*)d_in[21];
  const float* t_ln2_g     = (const float*)d_in[22];
  const float* t_ln2_b     = (const float*)d_in[23];
  const float* t_ff1_w     = (const float*)d_in[24];
  const float* t_ff1_b     = (const float*)d_in[25];
  const float* t_ff2_w     = (const float*)d_in[26];
  const float* t_ff2_b     = (const float*)d_in[27];
  const float* te1_w       = (const float*)d_in[28];
  const float* te1_b       = (const float*)d_in[29];
  const float* te_ln1_g    = (const float*)d_in[30];
  const float* te_ln1_b    = (const float*)d_in[31];
  const float* te2_w       = (const float*)d_in[32];
  const float* te2_b       = (const float*)d_in[33];
  const float* te_ln2_g    = (const float*)d_in[34];
  const float* te_ln2_b    = (const float*)d_in[35];
  const float* c_w         = (const float*)d_in[36];
  const float* c_b         = (const float*)d_in[37];
  const float* c_ln_g      = (const float*)d_in[38];
  const float* c_ln_b      = (const float*)d_in[39];

  char* ws = (char*)d_ws;
  size_t off = 0;
  auto alloc = [&](size_t bytes) {
    size_t o = off;
    off = (off + bytes + 255) & ~(size_t)255;
    return o;
  };

  // packed weights (frag-major f16, 1KB / fragment)
  const float* psrc[9] = {ia_in_w, gf_wi, gb_wi, gf_wh, gb_wh,
                          ta_in_w, ta_out_w, t_ff1_w, t_ff2_w};
  const int pN[9] = {192, 384, 384, 384, 384, 768, 256, 512, 256};
  const int pK[9] = {64, 64, 64, 128, 128, 256, 256, 256, 512};
  size_t poff[9];
  for (int i = 0; i < 9; ++i)
    poff[i] = alloc((size_t)(pK[i] / 32) * (pN[i] / 16) * 1024);

  const int BO = 51200;  // B*O
  size_t o_oe   = alloc((size_t)BO * 64 * sizeof(float));
  size_t o_g    = alloc((size_t)BO * 256 * sizeof(float));
  size_t o_qkv  = alloc((size_t)BO * 768 * sizeof(_Float16));  // also reused as h2 (f32, smaller)
  size_t o_ao   = alloc((size_t)BO * 256 * sizeof(_Float16));
  size_t o_h1   = alloc((size_t)BO * 256 * sizeof(float));

  float*    oe   = (float*)(ws + o_oe);
  float*    gbuf = (float*)(ws + o_g);
  _Float16* qkvT = (_Float16*)(ws + o_qkv);
  _Float16* aout = (_Float16*)(ws + o_ao);
  float*    h1   = (float*)(ws + o_h1);
  float*    h2   = (float*)(ws + o_qkv);  // alias: qkv dead after attention

  for (int i = 0; i < 9; ++i) {
    int nfrag = (pK[i] / 32) * (pN[i] / 16);
    int thr = nfrag * 32;
    pack_w<<<(thr + 255) / 256, 256, 0, stream>>>(
        psrc[i], (_Float16*)(ws + poff[i]), pN[i], pK[i]);
  }

  k_item_attn<<<BO, 128, 0, stream>>>(
      hist, item_emb, (const _Float16*)(ws + poff[0]), ia_in_b,
      ia_out_w, ia_out_b, oe);

  k_gru<<<dim3(64, 2), 256, 0, stream>>>(
      oe,
      (const _Float16*)(ws + poff[1]), (const _Float16*)(ws + poff[3]), gf_bi, gf_bh,
      (const _Float16*)(ws + poff[2]), (const _Float16*)(ws + poff[4]), gb_bi, gb_bh,
      gbuf);

  k_t_qkv<<<BO / 16, 256, 0, stream>>>(
      gbuf, (const _Float16*)(ws + poff[5]), ta_in_b, qkvT);

  k_t_attn<<<dim3(1024, 4), 256, 0, stream>>>(qkvT, aout);

  k_t_proj_ln<<<BO / 16, 256, 0, stream>>>(
      aout, (const _Float16*)(ws + poff[6]), ta_out_b, gbuf,
      t_ln1_g, t_ln1_b, h1);

  k_t_ffn<<<BO / 16, 256, 0, stream>>>(
      h1, (const _Float16*)(ws + poff[7]), t_ff1_b,
      (const _Float16*)(ws + poff[8]), t_ff2_b,
      t_ln2_g, t_ln2_b, h2);

  k_final<<<1024, 256, 0, stream>>>(
      h2, daysb, dayss,
      te1_w, te1_b, te_ln1_g, te_ln1_b,
      te2_w, te2_b, te_ln2_g, te_ln2_b,
      c_w, c_b, c_ln_g, c_ln_b,
      (float*)d_out);
}
